// Powerful_52467320488014
// MI455X (gfx1250) — compile-verified, matrix operand order checked by reading, not run on hardware
//
#include <hip/hip_runtime.h>

// ---------------------------------------------------------------------------
// PPGN forward for MI455X (gfx1250, wave32).
//
// Roofline: ~42 GFLOP total, largest tensor 75.5 MB (L2-resident at 192 MB),
// so compute-bound -> all big GEMMs via v_wmma_f32_16x16x32_bf16.
//  * m1/m2 linears -> WMMA, outputs scattered into bf16 layouts
//    o1A[b][c][i][k] and o2T[b][c][j][k].
//  * einsum 'bikc,bkjc->bijc' -> 512 independent 192^3 bf16 GEMMs.
//    Workgroup-cooperative: 4 waves per 64x64 tile, A/B K-slices staged to
//    LDS with GLOBAL_LOAD_ASYNC_TO_LDS_B128 (double-buffered, ASYNCcnt +
//    s_wait_asynccnt + barrier), fragments read back as ds_load_b128.
//  * m4 concat linear ([e||u] @ W4) -> WMMA with zero-padded K (96 / 128).
// Small ops (adj scatter, BN stats, trace/off-diag reductions, MLP heads,
// log-softmax) are scalar kernels (bandwidth-trivial).
//
// d_in flattening assumption (depth-first, dict insertion order):
//  0:x  1:np.lin1.w 2:np.lin1.b 3:np.lin2.w 4:np.lin3.w
//  5..10  conv0 {m1.w,m1.b,m2.w,m2.b,m4.w,m4.b}
//  11..16 conv1, 17..22 conv2
//  23..28 bn{0,1,2}{gamma,beta}
//  29..32 ext0 {lin1.w,lin1.b,lin2.w,lin3.w}, 33..36 ext1, 37..40 ext2
//  41:after_conv.w 42:after_conv.b 43:final.w 44:final.b 45:edge_index 46:batch
// ---------------------------------------------------------------------------

#define NB   8
#define NN   192
#define FINC 16
#define HID  64
#define HFW  128
#define NLAY 3
#define NCLS 10
#define RTOT (NB * NN * NN)   // 294912 rows

typedef __bf16 bf16_t;
typedef __attribute__((ext_vector_type(16))) __bf16        v16bf;
typedef __attribute__((ext_vector_type(8)))  float         v8f;
typedef __attribute__((ext_vector_type(4)))  unsigned int  u32x4;

union AF { v16bf v; u32x4 q[2]; unsigned short s[16]; };
union CF { v8f v; float f[8]; };

__device__ __forceinline__ unsigned short f32_to_bf16(float f) {
  unsigned int u = __float_as_uint(f);
  unsigned int r = u + 0x7FFFu + ((u >> 16) & 1u);   // round-to-nearest-even
  return (unsigned short)(r >> 16);
}

// ---- CDNA5 async global->LDS (ASYNCcnt-tracked), per-lane 16B transfers ----
__device__ __forceinline__ void async_ld16(unsigned int lds_off, unsigned long long gaddr) {
  asm volatile("global_load_async_to_lds_b128 %0, %1, off"
               :: "v"(lds_off), "v"(gaddr)
               : "memory");
}
__device__ __forceinline__ void wait_async0() {
  asm volatile("s_wait_asynccnt 0x0" ::: "memory");
}

// ------------------------------ small kernels ------------------------------

__global__ void k_zero_f32(float* p, long n) {
  long i = (long)blockIdx.x * blockDim.x + threadIdx.x;
  long stride = (long)gridDim.x * blockDim.x;
  for (; i < n; i += stride) p[i] = 0.f;
}

// adj scatter-add into channel 0 of u0 (B,N,N,17)
__global__ void k_add_edges(const int* __restrict__ ei, const int* __restrict__ batch,
                            float* __restrict__ u0, int nE) {
  int e = blockIdx.x * blockDim.x + threadIdx.x;
  if (e >= nE) return;
  int s = ei[e], d = ei[nE + e];
  int b = batch[s];
  int ps = s - b * NN;          // counts are uniform (=N) so starts[b]=b*N
  int pd = d - b * NN;
  atomicAdd(&u0[((size_t)(b * NN + ps) * NN + pd) * 17], 1.0f);
}

// diag-embed node features into channels 1..16 of u0
__global__ void k_set_diag(const float* __restrict__ x, const int* __restrict__ batch,
                           float* __restrict__ u0) {
  int idx = blockIdx.x * blockDim.x + threadIdx.x;   // 1536*16
  if (idx >= NB * NN * FINC) return;
  int node = idx >> 4, f = idx & 15;
  int b = batch[node];
  int p = node - b * NN;
  u0[((size_t)(b * NN + p) * NN + p) * 17 + 1 + f] = x[idx];
}

// W (K x 64) f32  ->  Wt (64 x Kp) bf16, zero padded
__global__ void k_prep_w(const float* __restrict__ W, int K, int Kp,
                         unsigned short* __restrict__ Wt) {
  int idx = blockIdx.x * blockDim.x + threadIdx.x;
  if (idx >= HID * Kp) return;
  int c = idx / Kp, k = idx - c * Kp;
  float v = (k < K) ? W[k * HID + c] : 0.f;
  Wt[idx] = f32_to_bf16(v);
}

// ------------------------- WMMA channel-mixing GEMM ------------------------
// Fill 8 elements (k = kb..kb+7) of one A-fragment half from the virtual concat
// [x1 | x2 | 0-pad]; vectorized 128-bit path when the chunk is fully inside
// one aligned source.
__device__ __forceinline__ void load_chunk(AF& a, int half,
                                           const float* __restrict__ x1, int K1,
                                           const float* __restrict__ x2, int K2,
                                           int row, int kb) {
  if (((K1 & 3) == 0) && (kb + 8 <= K1)) {
    const float* p = x1 + (size_t)row * K1 + kb;        // 16B aligned
    u32x4 f0 = *reinterpret_cast<const u32x4*>(p);
    u32x4 f1 = *reinterpret_cast<const u32x4*>(p + 4);
#pragma unroll
    for (int e = 0; e < 4; ++e) {
      a.s[half * 8 + e]     = f32_to_bf16(__uint_as_float(f0[e]));
      a.s[half * 8 + 4 + e] = f32_to_bf16(__uint_as_float(f1[e]));
    }
  } else if (x2 && ((K2 & 3) == 0) && (kb >= K1) && (kb - K1 + 8 <= K2)) {
    const float* p = x2 + (size_t)row * K2 + (kb - K1); // 16B aligned
    u32x4 f0 = *reinterpret_cast<const u32x4*>(p);
    u32x4 f1 = *reinterpret_cast<const u32x4*>(p + 4);
#pragma unroll
    for (int e = 0; e < 4; ++e) {
      a.s[half * 8 + e]     = f32_to_bf16(__uint_as_float(f0[e]));
      a.s[half * 8 + 4 + e] = f32_to_bf16(__uint_as_float(f1[e]));
    }
  } else {
#pragma unroll
    for (int e = 0; e < 8; ++e) {
      int k = kb + e;
      float v = 0.f;
      if (k < K1)                   v = x1[(size_t)row * K1 + k];
      else if (x2 && (k - K1) < K2) v = x2[(size_t)row * K2 + (k - K1)];
      a.s[half * 8 + e] = f32_to_bf16(v);
    }
  }
}

// O[r, 0..63] = concat(x1[r,:K1], x2[r,:K2], 0-pad)[0..Kp) @ Wt^T + bias
// mode 0: f32 row-major out (B*N*N, 64)
// mode 1: bf16 o1A[((b*64+c)*192 + i)*192 + k]   (r = (b,i,k))
// mode 2: bf16 o2T[((b*64+c)*192 + k)*192 + i]   (o2 transposed per (b,c))
__global__ void __launch_bounds__(256)
k_wmma_linear(const float* __restrict__ x1, int K1,
              const float* __restrict__ x2, int K2, int Kp,
              const unsigned short* __restrict__ Wt,
              const float* __restrict__ bias,
              void* __restrict__ outp, int mode) {
  int gid  = blockIdx.x * blockDim.x + threadIdx.x;
  int wave = gid >> 5, lane = gid & 31;
  int rbase = wave * 16;
  if (rbase >= RTOT) return;
  int lr = lane & 15, ksel = lane >> 4;
  int row = rbase + lr;                      // A-fragment row for this lane

  CF acc[4];
#pragma unroll
  for (int g = 0; g < 4; ++g)
#pragma unroll
    for (int v = 0; v < 8; ++v) acc[g].f[v] = 0.f;

  for (int k0 = 0; k0 < Kp; k0 += 32) {
    AF a;
    load_chunk(a, 0, x1, K1, x2, K2, row, k0 + 8 * ksel);
    load_chunk(a, 1, x1, K1, x2, K2, row, k0 + 16 + 8 * ksel);
#pragma unroll
    for (int g = 0; g < 4; ++g) {
      AF bf;
      const unsigned short* wp = Wt + (size_t)(g * 16 + lr) * Kp;
      bf.q[0] = *reinterpret_cast<const u32x4*>(wp + k0 + 8 * ksel);
      bf.q[1] = *reinterpret_cast<const u32x4*>(wp + k0 + 16 + 8 * ksel);
      acc[g].v = __builtin_amdgcn_wmma_f32_16x16x32_bf16(
          false, a.v, false, bf.v, (short)0, acc[g].v, false, false);
    }
  }

  // r = (b*N + i)*N + q ; rbase is 16-aligned and N%16==0 so tile shares (b,i)
  int q0 = rbase % NN;
  int rem = rbase / NN;
  int i = rem % NN;
  int b = rem / NN;

#pragma unroll
  for (int g = 0; g < 4; ++g) {
    int c = g * 16 + lr;
    float bi = bias ? bias[c] : 0.f;
#pragma unroll
    for (int v = 0; v < 8; ++v) {
      int rrow = v + 8 * ksel;
      float val = acc[g].f[v] + bi;
      if (mode == 0) {
        ((float*)outp)[(size_t)(rbase + rrow) * HID + c] = val;
      } else if (mode == 1) {
        ((unsigned short*)outp)[((size_t)(b * HID + c) * NN + i) * NN + q0 + rrow] =
            f32_to_bf16(val);
      } else {
        ((unsigned short*)outp)[((size_t)(b * HID + c) * NN + (q0 + rrow)) * NN + i] =
            f32_to_bf16(val);
      }
    }
  }
}

// ------------------------------ einsum (WMMA) ------------------------------
// E[b,i,j,c] = (1/sqrt(N)) * sum_k o1A[bc][i][k] * o2T[bc][j][k]
// One 128-thread block (4 waves) per 64x64 output tile of one (b,c) matrix.
// A/B 64x32 K-slices double-buffered in LDS via async-to-LDS.

__device__ __forceinline__ void store_tile(const CF& acc, float* __restrict__ E,
                                           int b, int c, int ibase, int jbase,
                                           int lr, int ksel, float scale) {
#pragma unroll
  for (int v = 0; v < 8; ++v) {
    int i = ibase + v + 8 * ksel;
    int j = jbase + lr;
    E[((size_t)(b * NN + i) * NN + j) * HID + c] = acc.f[v] * scale;
  }
}

__global__ void __launch_bounds__(128)
k_einsum_wmma(const unsigned short* __restrict__ Abuf,
              const unsigned short* __restrict__ Bbuf,
              float* __restrict__ Ebuf) {
  __shared__ alignas(16) unsigned short As[2][64][32];
  __shared__ alignas(16) unsigned short Bs[2][64][32];

  int bid = blockIdx.x;                       // 512 * 9 blocks
  int bc = bid / 9;
  int t  = bid - bc * 9;
  int TI = (t / 3) * 64, TJ = (t % 3) * 64;
  const unsigned short* A = Abuf + (size_t)bc * NN * NN;
  const unsigned short* B = Bbuf + (size_t)bc * NN * NN;

  int tid  = threadIdx.x;
  int wave = tid >> 5, lane = tid & 31;
  int lr = lane & 15, ksel = lane >> 4;
  int li = (wave >> 1) * 32, lj = (wave & 1) * 32;   // quadrant within tile

  // async stage of one 64x32 K-slice of A and B: 256 16B chunks each,
  // 2 chunks per source per thread (each call covers a full wave = 512B).
  auto issue_async = [&](int bf, int k0) {
#pragma unroll
    for (int c = 0; c < 2; ++c) {
      int ch = tid * 2 + c;
      int r = ch >> 2, q = ch & 3;            // row 0..63, 16B quarter 0..3
      unsigned long long ga =
          (unsigned long long)(uintptr_t)(A + (size_t)(TI + r) * NN + k0 + q * 8);
      unsigned long long gb =
          (unsigned long long)(uintptr_t)(B + (size_t)(TJ + r) * NN + k0 + q * 8);
      async_ld16((unsigned int)(uintptr_t)&As[bf][r][q * 8], ga);
      async_ld16((unsigned int)(uintptr_t)&Bs[bf][r][q * 8], gb);
    }
  };

  CF c00, c01, c10, c11;
#pragma unroll
  for (int v = 0; v < 8; ++v) { c00.f[v] = c01.f[v] = c10.f[v] = c11.f[v] = 0.f; }

  issue_async(0, 0);
  int bf = 0;
  for (int s = 0; s < NN / 32; ++s) {
    wait_async0();            // own wave's async loads landed in LDS
    __syncthreads();          // all waves' slices visible; prev buffer free
    if (s + 1 < NN / 32) issue_async(bf ^ 1, (s + 1) * 32);

    AF a0, a1, b0, b1;
    int o0 = 8 * ksel, o1 = 16 + 8 * ksel;
    a0.q[0] = *reinterpret_cast<const u32x4*>(&As[bf][li + lr][o0]);
    a0.q[1] = *reinterpret_cast<const u32x4*>(&As[bf][li + lr][o1]);
    a1.q[0] = *reinterpret_cast<const u32x4*>(&As[bf][li + 16 + lr][o0]);
    a1.q[1] = *reinterpret_cast<const u32x4*>(&As[bf][li + 16 + lr][o1]);
    b0.q[0] = *reinterpret_cast<const u32x4*>(&Bs[bf][lj + lr][o0]);
    b0.q[1] = *reinterpret_cast<const u32x4*>(&Bs[bf][lj + lr][o1]);
    b1.q[0] = *reinterpret_cast<const u32x4*>(&Bs[bf][lj + 16 + lr][o0]);
    b1.q[1] = *reinterpret_cast<const u32x4*>(&Bs[bf][lj + 16 + lr][o1]);

    c00.v = __builtin_amdgcn_wmma_f32_16x16x32_bf16(false, a0.v, false, b0.v, (short)0, c00.v, false, false);
    c01.v = __builtin_amdgcn_wmma_f32_16x16x32_bf16(false, a0.v, false, b1.v, (short)0, c01.v, false, false);
    c10.v = __builtin_amdgcn_wmma_f32_16x16x32_bf16(false, a1.v, false, b0.v, (short)0, c10.v, false, false);
    c11.v = __builtin_amdgcn_wmma_f32_16x16x32_bf16(false, a1.v, false, b1.v, (short)0, c11.v, false, false);

    bf ^= 1;
  }

  const float scale = 0.07216878364870323f;   // 1/sqrt(192)
  int b = bc >> 6, c = bc & 63;
  int wi = TI + li, wj = TJ + lj;
  store_tile(c00, Ebuf, b, c, wi,      wj,      lr, ksel, scale);
  store_tile(c01, Ebuf, b, c, wi,      wj + 16, lr, ksel, scale);
  store_tile(c10, Ebuf, b, c, wi + 16, wj,      lr, ksel, scale);
  store_tile(c11, Ebuf, b, c, wi + 16, wj + 16, lr, ksel, scale);
}

// ------------------------------- batch norm --------------------------------

__global__ void k_bn_stats(const float* __restrict__ X, float* __restrict__ mean,
                           float* __restrict__ var) {
  __shared__ float s1[256], s2[256];
  int c = blockIdx.x, t = threadIdx.x;
  float a = 0.f, b = 0.f;
  for (int r = t; r < RTOT; r += 256) {
    float v = X[(size_t)r * HID + c];
    a += v; b += v * v;
  }
  s1[t] = a; s2[t] = b; __syncthreads();
  for (int s = 128; s > 0; s >>= 1) {
    if (t < s) { s1[t] += s1[t + s]; s2[t] += s2[t + s]; }
    __syncthreads();
  }
  if (t == 0) {
    float m = s1[0] / (float)RTOT;
    mean[c] = m;
    var[c]  = s2[0] / (float)RTOT - m * m;
  }
}

__global__ void k_bn_apply(const float* __restrict__ X, float* __restrict__ Y,
                           const float* __restrict__ mean, const float* __restrict__ var,
                           const float* __restrict__ gamma, const float* __restrict__ beta) {
  int idx = blockIdx.x * blockDim.x + threadIdx.x;
  if (idx >= RTOT * HID) return;
  int c = idx & (HID - 1);
  float rs = __frsqrt_rn(var[c] + 1e-5f);
  Y[idx] = (X[idx] - mean[c]) * rs * gamma[c] + beta[c];
}

// --------------------------- feature extractor -----------------------------

__global__ void k_fe_stats(const float* __restrict__ X, int C,
                           float* __restrict__ tr, float* __restrict__ tot) {
  __shared__ float s1[256], s2[256];
  int blk = blockIdx.x;
  int b = blk / C, c = blk - b * C;
  int t = threadIdx.x;
  const float* base = X + (size_t)b * NN * NN * C + c;
  float a = 0.f, d = 0.f;
  for (int ij = t; ij < NN * NN; ij += 256) a += base[(size_t)ij * C];
  for (int i = t; i < NN; i += 256)         d += base[(size_t)(i * NN + i) * C];
  s1[t] = a; s2[t] = d; __syncthreads();
  for (int s = 128; s > 0; s >>= 1) {
    if (t < s) { s1[t] += s1[t + s]; s2[t] += s2[t + s]; }
    __syncthreads();
  }
  if (t == 0) { tot[blk] = s1[0]; tr[blk] = s2[0]; }
}

__global__ void k_fe_apply(const float* __restrict__ tr, const float* __restrict__ tot, int C,
                           const float* __restrict__ w1, const float* __restrict__ b1,
                           const float* __restrict__ w2, const float* __restrict__ w3,
                           float* __restrict__ out, int init) {
  int b = blockIdx.x, h = threadIdx.x;   // 128 threads
  __shared__ float o[HFW];
  const float n = (float)NN;
  float o1 = b1[h], o2 = 0.f;
  for (int k = 0; k < C; ++k) {
    float trv = tr[b * C + k];
    float tv  = trv / n;
    float sv  = (tot[b * C + k] - trv) / (n * (n - 1.f));
    o1 += tv * w1[k * HFW + h];
    o2 += sv * w2[k * HFW + h];
  }
  float v = o1 + o2;
  o[h] = v; __syncthreads();
  float acc = v;
  for (int k = 0; k < HFW; ++k) {
    float z = o[k];
    z = z >= 0.f ? z : 0.01f * z;       // leaky relu
    acc += z * w3[k * HFW + h];
  }
  if (init) out[b * HFW + h] = acc;
  else      out[b * HFW + h] += acc;
}

// ---------------------------------- head -----------------------------------

__global__ void k_head(const float* __restrict__ outacc,
                       const float* __restrict__ wac, const float* __restrict__ bac,
                       const float* __restrict__ wf, const float* __restrict__ bfv,
                       float* __restrict__ dout) {
  int b = blockIdx.x, h = threadIdx.x;   // 128 threads
  __shared__ float o[HFW], a[HFW], lg[NCLS], red[2];
  float v = outacc[b * HFW + h];
  v = fmaxf(v, 0.f) / (float)NLAY;
  o[h] = v; __syncthreads();
  float s = bac[h];
  for (int k = 0; k < HFW; ++k) s += o[k] * wac[k * HFW + h];
  a[h] = v + fmaxf(s, 0.f);
  __syncthreads();
  if (h < NCLS) {
    float L = bfv[h];
    for (int k = 0; k < HFW; ++k) L += a[k] * wf[k * NCLS + h];
    lg[h] = L;
  }
  __syncthreads();
  if (h == 0) {
    float mx = lg[0];
    for (int k = 1; k < NCLS; ++k) mx = fmaxf(mx, lg[k]);
    float se = 0.f;
    for (int k = 0; k < NCLS; ++k) se += __expf(lg[k] - mx);
    red[0] = mx; red[1] = __logf(se);
  }
  __syncthreads();
  if (h < NCLS) dout[b * NCLS + h] = lg[h] - red[0] - red[1];
}

// ------------------------------ host launcher ------------------------------

extern "C" void kernel_launch(void* const* d_in, const int* in_sizes, int n_in,
                              void* d_out, int out_size, void* d_ws, size_t ws_size,
                              hipStream_t stream) {
  const float* x        = (const float*)d_in[0];
  const float* np_w1    = (const float*)d_in[1];
  const float* np_b1    = (const float*)d_in[2];
  const float* np_w2    = (const float*)d_in[3];
  const float* np_w3    = (const float*)d_in[4];
  const float* cm_w[NLAY][3]; const float* cm_b[NLAY][3];
  for (int l = 0; l < NLAY; ++l) {
    int base = 5 + l * 6;
    cm_w[l][0] = (const float*)d_in[base + 0]; cm_b[l][0] = (const float*)d_in[base + 1];
    cm_w[l][1] = (const float*)d_in[base + 2]; cm_b[l][1] = (const float*)d_in[base + 3];
    cm_w[l][2] = (const float*)d_in[base + 4]; cm_b[l][2] = (const float*)d_in[base + 5];
  }
  const float* bn_g[NLAY]; const float* bn_b[NLAY];
  for (int l = 0; l < NLAY; ++l) {
    bn_g[l] = (const float*)d_in[23 + 2 * l];
    bn_b[l] = (const float*)d_in[24 + 2 * l];
  }
  const float* ex_w1[NLAY]; const float* ex_b1[NLAY];
  const float* ex_w2[NLAY]; const float* ex_w3[NLAY];
  for (int l = 0; l < NLAY; ++l) {
    int base = 29 + l * 4;
    ex_w1[l] = (const float*)d_in[base + 0];
    ex_b1[l] = (const float*)d_in[base + 1];
    ex_w2[l] = (const float*)d_in[base + 2];
    ex_w3[l] = (const float*)d_in[base + 3];
  }
  const float* ac_w  = (const float*)d_in[41];
  const float* ac_b  = (const float*)d_in[42];
  const float* fin_w = (const float*)d_in[43];
  const float* fin_b = (const float*)d_in[44];
  const int* edge_index = (const int*)d_in[45];
  const int* batch      = (const int*)d_in[46];
  int nE = in_sizes[45] / 2;

  // workspace partition (peak ~323 MB)
  char* ws = (char*)d_ws;
  size_t off = 0;
  auto alloc = [&](size_t bytes) -> void* {
    off = (off + 255) & ~(size_t)255;
    void* p = ws + off;
    off += bytes;
    return p;
  };
  float* u0  = (float*)alloc((size_t)RTOT * 17 * 4);
  float* u   = (float*)alloc((size_t)RTOT * HID * 4);
  float* tmp = (float*)alloc((size_t)RTOT * HID * 4);
  float* e   = (float*)alloc((size_t)RTOT * HID * 4);
  unsigned short* o1A = (unsigned short*)alloc((size_t)NB * HID * NN * NN * 2);
  unsigned short* o2T = (unsigned short*)alloc((size_t)NB * HID * NN * NN * 2);
  unsigned short* wt1 = (unsigned short*)alloc(HID * 64 * 2);
  unsigned short* wt2 = (unsigned short*)alloc(HID * 64 * 2);
  unsigned short* wt4 = (unsigned short*)alloc(HID * 128 * 2);
  float* meanb  = (float*)alloc(HID * 4);
  float* varb   = (float*)alloc(HID * 4);
  float* trb    = (float*)alloc(NB * HID * 4);
  float* totb   = (float*)alloc(NB * HID * 4);
  float* outacc = (float*)alloc(NB * HFW * 4);

  const int gemmWaves = RTOT / 16;                 // 18432
  dim3 gemmGrid(gemmWaves * 32 / 256), blk256(256);
  dim3 einGrid(NB * HID * 9), blk128(128);         // 4608 blocks, 4 waves each

  // build dense u0 = concat(adj, diag(x))
  k_zero_f32<<<2048, 256, 0, stream>>>(u0, (long)RTOT * 17);
  k_add_edges<<<(nE + 255) / 256, 256, 0, stream>>>(edge_index, batch, u0, nE);
  k_set_diag<<<(NB * NN * FINC + 255) / 256, 256, 0, stream>>>(x, batch, u0);

  // no_prop feature extractor -> out (init)
  k_fe_stats<<<NB * 17, 256, 0, stream>>>(u0, 17, trb, totb);
  k_fe_apply<<<NB, HFW, 0, stream>>>(trb, totb, 17, np_w1, np_b1, np_w2, np_w3, outacc, 1);

  for (int l = 0; l < NLAY; ++l) {
    int Cin = (l == 0) ? 17 : HID;
    int Kp  = (l == 0) ? 32 : 64;
    int K4  = HID + Cin;
    int Kp4 = (l == 0) ? 96 : 128;
    const float* Xu = (l == 0) ? u0 : u;

    k_prep_w<<<(HID * Kp  + 255) / 256, 256, 0, stream>>>(cm_w[l][0], Cin, Kp,  wt1);
    k_prep_w<<<(HID * Kp  + 255) / 256, 256, 0, stream>>>(cm_w[l][1], Cin, Kp,  wt2);
    k_prep_w<<<(HID * Kp4 + 255) / 256, 256, 0, stream>>>(cm_w[l][2], K4,  Kp4, wt4);

    // o1 = (u@W1+b1) -> A-layout bf16 ; o2 = (u@W2+b2) -> transposed bf16
    k_wmma_linear<<<gemmGrid, blk256, 0, stream>>>(Xu, Cin, nullptr, 0, Kp, wt1, cm_b[l][0], o1A, 1);
    k_wmma_linear<<<gemmGrid, blk256, 0, stream>>>(Xu, Cin, nullptr, 0, Kp, wt2, cm_b[l][1], o2T, 2);

    // 512 independent 192x192x192 bf16 GEMMs (async-LDS staged)
    k_einsum_wmma<<<einGrid, blk128, 0, stream>>>(o1A, o2T, e);

    // m4 on concat([e, u]) -> tmp (f32)
    k_wmma_linear<<<gemmGrid, blk256, 0, stream>>>(e, HID, Xu, Cin, Kp4, wt4, cm_b[l][2], tmp, 0);

    // batch norm (training-mode stats) -> u
    k_bn_stats<<<HID, 256, 0, stream>>>(tmp, meanb, varb);
    k_bn_apply<<<(RTOT * HID + 255) / 256, 256, 0, stream>>>(tmp, u, meanb, varb, bn_g[l], bn_b[l]);

    // extractor -> accumulate into out
    k_fe_stats<<<NB * HID, 256, 0, stream>>>(u, HID, trb, totb);
    k_fe_apply<<<NB, HFW, 0, stream>>>(trb, totb, HID, ex_w1[l], ex_b1[l], ex_w2[l], ex_w3[l], outacc, 0);
  }

  // relu/NL, after_conv residual, final linear, log_softmax
  k_head<<<NB, HFW, 0, stream>>>(outacc, ac_w, ac_b, fin_w, fin_b, (float*)d_out);

  (void)n_in; (void)out_size; (void)ws_size;
}
